// NormalGuidedDeformTransformerBlock1D_82437602280007
// MI455X (gfx1250) — matrix operand order repaired and checked by reading, not compiled, not run on hardware
//
#include <hip/hip_runtime.h>
#include <cmath>

// ---------------- problem constants ----------------
#define B2   4            // 2*B concatenated batches
#define Cc   128          // channels
#define Hh   64
#define Wd   256
#define HW   (Hh * Wd)    // 16384 pixels per batch image
#define Gg   8            // groups
#define Pp   4            // points per branch
#define Dd   16           // C / G
#define KSZ  128          // K of every GEMM pass (256-K is split)
#define LPAD 8            // LDS row padding in halves (16B, keeps v8h alignment)
#define TPB  (HW / 16)    // 16-pixel M-tiles per batch = 1024

typedef __attribute__((ext_vector_type(16))) _Float16 v16h;
typedef __attribute__((ext_vector_type(8)))  _Float16 v8h;
typedef __attribute__((ext_vector_type(8)))  float    v8f;

__device__ __forceinline__ float fast_rcp(float x) {
    return __builtin_amdgcn_rcpf(x);      // v_rcp_f32
}

// =====================================================================
// Generic per-pixel 1x1-conv GEMM:  Out[b][n][hw] = act( W @ A + bias )
//   A   : (nBatchLocal, KSZ, HW) f32, channel-major (k stride = HW)
//   Wm  : (Nout, wStride) f32 row-major, K contiguous within a row
//   Out : (B2, outStride, HW) f32
// One wave32 owns one 16-pixel M-tile and the full N range.
// WMMA: v_wmma_f32_16x16x32_f16, f32 accumulation.
// ACT: 0 = none, 1 = SiLU.  ACCUM: 0 = init from bias, 1 = add to Out.
// =====================================================================
template <int ACT, int ACCUM>
__global__ __launch_bounds__(128)
void gemm128_f16(const float* __restrict__ A,
                 const float* __restrict__ Wm,
                 const float* __restrict__ bias,
                 float* __restrict__ Out,
                 int Nout, int wStride, int outStride,
                 int outChanOff, int outBatchOff)
{
    __shared__ _Float16 w_lds[Cc * (KSZ + LPAD)];          // up to 128x128 f16 weights
    __shared__ _Float16 a_lds[4][16][KSZ + LPAD];          // one 16xK tile per wave

    const int lane = threadIdx.x & 31;
    const int wave = threadIdx.x >> 5;

    // warm GL2 for the weight rows this block is about to stage
    __builtin_prefetch(Wm + (size_t)threadIdx.x * wStride, 0, 1);

    // ---- stage weights f32 -> f16 (block cooperative, coalesced) ----
    for (int i = threadIdx.x; i < Nout * KSZ; i += 128) {
        int n = i >> 7;          // / 128
        int k = i & 127;
        w_lds[n * (KSZ + LPAD) + k] = (_Float16)Wm[(size_t)n * wStride + k];
    }

    // ---- stage A tile f32 -> f16 (per-wave, coalesced along pixels) ----
    const int tile = blockIdx.x * 4 + wave;
    const int b    = tile / TPB;                // local batch index for A
    const int hw0  = (tile % TPB) * 16;
    const float* Ab = A + (size_t)b * KSZ * HW + hw0;
    for (int i = lane; i < 16 * KSZ; i += 32) {
        int k = i >> 4;
        int m = i & 15;
        a_lds[wave][m][k] = (_Float16)Ab[(size_t)k * HW + m];
    }
    __syncthreads();

    const int halfSel = lane >> 4;              // 0: lanes 0-15, 1: lanes 16-31
    const int lid     = lane & 15;
    const _Float16* arow = &a_lds[wave][lid][0];
    const int bOut = b + outBatchOff;

    for (int nt = 0; nt < (Nout >> 4); ++nt) {
        const int n = nt * 16 + lid;
        float* orow = Out + (size_t)bOut * outStride * HW
                          + (size_t)(outChanOff + n) * HW + hw0 + 8 * halfSel;
        v8f acc;
        if (ACCUM) {
            #pragma unroll
            for (int r = 0; r < 8; ++r) acc[r] = orow[r];
        } else {
            float bn = bias ? bias[n] : 0.0f;
            #pragma unroll
            for (int r = 0; r < 8; ++r) acc[r] = bn;
        }

        const _Float16* wr = &w_lds[n * (KSZ + LPAD)];
        #pragma unroll
        for (int kb = 0; kb < KSZ; kb += 32) {
            // A fragment (ISA 16-bit A 16x32 layout):
            //   e<8  -> K = kb + 8*halfSel + e
            //   e>=8 -> K = kb + 16 + 8*halfSel + (e-8)
            v8h a0 = *(const v8h*)(arow + kb +      8 * halfSel);
            v8h a1 = *(const v8h*)(arow + kb + 16 + 8 * halfSel);
            // B fragment: lane = N, same K layout as A
            v8h b0 = *(const v8h*)(wr + kb +      8 * halfSel);
            v8h b1 = *(const v8h*)(wr + kb + 16 + 8 * halfSel);
            v16h av, bv;
            #pragma unroll
            for (int e = 0; e < 8; ++e) {
                av[e] = a0[e]; av[e + 8] = a1[e];
                bv[e] = b0[e]; bv[e + 8] = b1[e];
            }
            acc = __builtin_amdgcn_wmma_f32_16x16x32_f16(
                false, av, false, bv, (short)0, acc, false, false);
        }

        // epilogue: D layout lane=N, row m = r + 8*halfSel (folded into orow)
        #pragma unroll
        for (int r = 0; r < 8; ++r) {
            float v = acc[r];
            if (ACT == 1) {
                // SiLU = v * sigmoid(v); cheap v_rcp instead of IEEE divide
                v = v * fast_rcp(1.0f + __expf(-v));
            }
            orow[r] = v;
        }
    }
}

// =====================================================================
// nfeat = w_normal @ geo + b_normal   (K=3, pure VALU; geo = [nr; nl])
// =====================================================================
__global__ void nfeat_kernel(const float* __restrict__ nl, const float* __restrict__ nr,
                             const float* __restrict__ wn, const float* __restrict__ bn,
                             float* __restrict__ outF, int total)
{
    int id = blockIdx.x * blockDim.x + threadIdx.x;
    if (id >= total) return;
    int hw = id % HW;
    int c  = (id / HW) % Cc;
    int b  = id / (Cc * HW);
    const float* g = (b < 2) ? (nr + (size_t)b * 3 * HW)
                             : (nl + (size_t)(b - 2) * 3 * HW);
    outF[id] = bn[c] + wn[c * 3 + 0] * g[hw]
                     + wn[c * 3 + 1] * g[HW + hw]
                     + wn[c * 3 + 2] * g[2 * HW + hw];
}

// =====================================================================
// Softmax over 2P attn logits, keypoints (written to d_out), 1D linear
// sampling of value/nfeat along W, attention-weighted aggregation.
// One thread per (batch, group, pixel).
// =====================================================================
__global__ void sample_kernel(const float* __restrict__ qp,    // (B2,128,HW): offc|offg|attn
                              const float* __restrict__ value, // (B2,C,HW)
                              const float* __restrict__ nfeat, // (B2,C,HW)
                              const float* __restrict__ cap,   // (B2,G,P,HW)
                              const float* __restrict__ gap,
                              float* __restrict__ agg,         // (B2,C,HW)
                              float* __restrict__ ckp_out,
                              float* __restrict__ gkp_out)
{
    int id = blockIdx.x * blockDim.x + threadIdx.x;
    if (id >= B2 * Gg * HW) return;
    int hw = id % HW;
    int g  = (id / HW) % Gg;
    int b  = id / (Gg * HW);
    int rowBase = hw - (hw % Wd);               // h * Wd

    const size_t qbase = (size_t)b * Cc * HW + hw;

    // softmax over the group's 2P = 8 logits (channels 64 + g*8 + j)
    float lg[8], mx = -1e30f;
    #pragma unroll
    for (int j = 0; j < 8; ++j) {
        lg[j] = qp[qbase + (size_t)(64 + g * 8 + j) * HW];
        mx = fmaxf(mx, lg[j]);
    }
    float s = 0.f;
    #pragma unroll
    for (int j = 0; j < 8; ++j) { lg[j] = __expf(lg[j] - mx); s += lg[j]; }
    float inv = fast_rcp(s);
    #pragma unroll
    for (int j = 0; j < 8; ++j) lg[j] *= inv;

    float acc[Dd];
    #pragma unroll
    for (int d = 0; d < Dd; ++d) acc[d] = 0.f;

    const size_t vbase = (size_t)b * Cc * HW + (size_t)(g * Dd) * HW + rowBase;

    for (int p = 0; p < Pp; ++p) {
        size_t aidx = (size_t)((b * Gg + g) * Pp + p) * HW + hw;

        // ---- context branch: sample `value` at ctx_kp ----
        float offc = qp[qbase + (size_t)(g * Pp + p) * HW];
        float x = cap[aidx] * (float)(Wd - 1) + offc;
        x = fminf(fmaxf(x, 0.f), (float)(Wd - 1));
        ckp_out[aidx] = x;
        float x0 = floorf(x);
        int   i0 = (int)x0;
        int   i1 = min(i0 + 1, Wd - 1);
        float t  = x - x0;
        float wc = lg[p];
        #pragma unroll
        for (int d = 0; d < Dd; ++d) {
            const float* vp = value + vbase + (size_t)d * HW;
            float v0 = vp[i0], v1 = vp[i1];
            acc[d] += wc * (v0 + (v1 - v0) * t);
        }

        // ---- geometric branch: sample `nfeat` at geo_kp ----
        float offg = qp[qbase + (size_t)(32 + g * Pp + p) * HW];
        float y = gap[aidx] * (float)(Wd - 1) + offg;
        y = fminf(fmaxf(y, 0.f), (float)(Wd - 1));
        gkp_out[aidx] = y;
        float y0 = floorf(y);
        int   j0 = (int)y0;
        int   j1 = min(j0 + 1, Wd - 1);
        float u  = y - y0;
        float wg = lg[Pp + p];
        #pragma unroll
        for (int d = 0; d < Dd; ++d) {
            const float* np = nfeat + vbase + (size_t)d * HW;
            float v0 = np[j0], v1 = np[j1];
            acc[d] += wg * (v0 + (v1 - v0) * u);
        }
    }

    #pragma unroll
    for (int d = 0; d < Dd; ++d)
        agg[(size_t)b * Cc * HW + (size_t)(g * Dd + d) * HW + hw] = acc[d];
}

// =====================================================================
// x = q + upd*mask ; LayerNorm over channels. In-place in `xupd`
// (reads upd, overwrites with normalized x). One thread per pixel.
// =====================================================================
__global__ void ln_kernel(const float* __restrict__ ml, const float* __restrict__ mr,
                          const float* __restrict__ gml, const float* __restrict__ gmr,
                          float* __restrict__ xupd,
                          const float* __restrict__ lnw, const float* __restrict__ lnb)
{
    int id = blockIdx.x * blockDim.x + threadIdx.x;
    if (id >= B2 * HW) return;
    int hw = id % HW;
    int b  = id / HW;
    const float* q = (b < 2) ? (ml + (size_t)b * Cc * HW)
                             : (mr + (size_t)(b - 2) * Cc * HW);
    float mk = (b < 2) ? gml[(size_t)b * HW + hw]
                       : gmr[(size_t)(b - 2) * HW + hw];
    size_t base = (size_t)b * Cc * HW + hw;

    float s = 0.f, ss = 0.f;
    for (int c = 0; c < Cc; ++c) {
        size_t o = base + (size_t)c * HW;
        float x = q[(size_t)c * HW + hw] + xupd[o] * mk;
        xupd[o] = x;
        s += x; ss += x * x;
    }
    float mu   = s * (1.0f / (float)Cc);     // exact: Cc is a power of two
    float var  = ss * (1.0f / (float)Cc) - mu * mu;
    float rinv = rsqrtf(var + 1e-5f);
    for (int c = 0; c < Cc; ++c) {
        size_t o = base + (size_t)c * HW;
        xupd[o] = (xupd[o] - mu) * rinv * lnw[c] + lnb[c];
    }
}

// =====================================================================
// Final residual + split back into ml / mr regions of d_out.
// =====================================================================
__global__ void final_kernel(const float* __restrict__ xn, const float* __restrict__ t,
                             float* __restrict__ out, int total)
{
    int id = blockIdx.x * blockDim.x + threadIdx.x;
    if (id >= total) return;
    int hw = id % HW;
    int c  = (id / HW) % Cc;
    int b  = id / (Cc * HW);
    float v = xn[id] + t[id];
    size_t o = (b < 2 ? (size_t)0 : (size_t)2 * Cc * HW)
             + (size_t)((b & 1) * Cc + c) * HW + hw;
    out[o] = v;
}

// =====================================================================
extern "C" void kernel_launch(void* const* d_in, const int* in_sizes, int n_in,
                              void* d_out, int out_size, void* d_ws, size_t ws_size,
                              hipStream_t stream)
{
    const float* match_left  = (const float*)d_in[0];
    const float* match_right = (const float*)d_in[1];
    const float* feat_left   = (const float*)d_in[2];
    const float* feat_right  = (const float*)d_in[3];
    const float* normal_left = (const float*)d_in[4];
    const float* normal_right= (const float*)d_in[5];
    const float* gml         = (const float*)d_in[6];
    const float* gmr         = (const float*)d_in[7];
    const float* cap         = (const float*)d_in[8];
    const float* gap         = (const float*)d_in[9];
    const float* w_value = (const float*)d_in[10];
    const float* b_value = (const float*)d_in[11];
    const float* w_normal= (const float*)d_in[12];
    const float* b_normal= (const float*)d_in[13];
    const float* w_offc  = (const float*)d_in[14];
    const float* b_offc  = (const float*)d_in[15];
    const float* w_offg  = (const float*)d_in[16];
    const float* b_offg  = (const float*)d_in[17];
    const float* w_attn  = (const float*)d_in[18];
    const float* b_attn  = (const float*)d_in[19];
    const float* w_out   = (const float*)d_in[20];
    const float* b_out   = (const float*)d_in[21];
    const float* ln_w    = (const float*)d_in[22];
    const float* ln_b    = (const float*)d_in[23];
    const float* w_ff1   = (const float*)d_in[24];
    const float* b_ff1   = (const float*)d_in[25];
    const float* w_ff2   = (const float*)d_in[26];
    const float* b_ff2   = (const float*)d_in[27];

    float* out = (float*)d_out;

    // 4 workspace slots of (B2,C,HW) f32 = 32 MB each (128 MB total)
    const size_t slot = (size_t)B2 * Cc * HW;
    float* S0 = (float*)d_ws;         // value, later h1a (silu)
    float* S1 = S0 + slot;            // nfeat, later h1b (silu)
    float* S2 = S1 + slot;            // qp -> upd -> xn (in place)
    float* S3 = S2 + slot;            // agg -> ffn output

    const dim3 blk(128);
    const int grid2 = (2 * TPB) / 4;  // two batches, 4 waves/block
    const int grid4 = (4 * TPB) / 4;  // four batches
    const int tot   = B2 * Cc * HW;

    // ---- value = wv @ ctx + bv ; ctx = [feat_right; feat_left] ----
    gemm128_f16<0,0><<<grid2, blk, 0, stream>>>(feat_right, w_value, b_value, S0, 128, 128, 128, 0, 0);
    gemm128_f16<0,0><<<grid2, blk, 0, stream>>>(feat_left,  w_value, b_value, S0, 128, 128, 128, 0, 2);

    // ---- nfeat = wn @ geo + bn ; geo = [normal_right; normal_left] ----
    nfeat_kernel<<<(tot + 255) / 256, 256, 0, stream>>>(normal_left, normal_right,
                                                        w_normal, b_normal, S1, tot);

    // ---- q projections -> S2 channels [0:32)=off_c [32:64)=off_g [64:128)=attn ----
    gemm128_f16<0,0><<<grid2, blk, 0, stream>>>(match_left,  w_offc, b_offc, S2, 32, 128, 128,  0, 0);
    gemm128_f16<0,0><<<grid2, blk, 0, stream>>>(match_right, w_offc, b_offc, S2, 32, 128, 128,  0, 2);
    gemm128_f16<0,0><<<grid2, blk, 0, stream>>>(match_left,  w_offg, b_offg, S2, 32, 128, 128, 32, 0);
    gemm128_f16<0,0><<<grid2, blk, 0, stream>>>(match_right, w_offg, b_offg, S2, 32, 128, 128, 32, 2);
    gemm128_f16<0,0><<<grid2, blk, 0, stream>>>(match_left,  w_attn, b_attn, S2, 64, 128, 128, 64, 0);
    gemm128_f16<0,0><<<grid2, blk, 0, stream>>>(match_right, w_attn, b_attn, S2, 64, 128, 128, 64, 2);

    // ---- softmax + keypoints + deformable sampling + aggregation ----
    float* ckp = out + (size_t)2 * 2 * Cc * HW;               // after ml|mr
    float* gkp = ckp + (size_t)B2 * Gg * Pp * HW;
    int sThreads = B2 * Gg * HW;
    sample_kernel<<<(sThreads + 255) / 256, 256, 0, stream>>>(S2, S0, S1, cap, gap,
                                                              S3, ckp, gkp);

    // ---- upd = wo @ agg + bo  (qp in S2 is dead -> overwrite) ----
    gemm128_f16<0,0><<<grid4, blk, 0, stream>>>(S3, w_out, b_out, S2, 128, 128, 128, 0, 0);

    // ---- x = q + upd*mask ; LayerNorm (in place in S2) ----
    int pix = B2 * HW;
    ln_kernel<<<(pix + 255) / 256, 256, 0, stream>>>(match_left, match_right,
                                                     gml, gmr, S2, ln_w, ln_b);

    // ---- FFN: h1 = silu(w1 @ xn + b1), N=256 split into two N=128 passes ----
    gemm128_f16<1,0><<<grid4, blk, 0, stream>>>(S2, w_ff1,             b_ff1,       S0, 128, 128, 128, 0, 0);
    gemm128_f16<1,0><<<grid4, blk, 0, stream>>>(S2, w_ff1 + 128 * 128, b_ff1 + 128, S1, 128, 128, 128, 0, 0);

    // ---- t = w2 @ h1 + b2, K=256 split into two K=128 accumulating passes ----
    gemm128_f16<0,0><<<grid4, blk, 0, stream>>>(S0, w_ff2,       b_ff2, S3, 128, 256, 128, 0, 0);
    gemm128_f16<0,1><<<grid4, blk, 0, stream>>>(S1, w_ff2 + 128, b_ff2, S3, 128, 256, 128, 0, 0);

    // ---- x = xn + t -> ml | mr in d_out ----
    final_kernel<<<(tot + 255) / 256, 256, 0, stream>>>(S2, S3, out, tot);
}
